// BioinspiredNeuralNetwork_70866960384008
// MI455X (gfx1250) — matrix operand order repaired and checked by reading, not compiled, not run on hardware
//
#include <hip/hip_runtime.h>
#include <stdint.h>

// BioinspiredNeuralNetwork for MI455X (gfx1250, wave32, WMMA bf16 + async-to-LDS)
//
// 3 x complex GEMM [512x4096]x[4096x4096] -> fused activation/mask.
// Compute-bound at bf16 (206 GFLOP vs 17us HBM floor): stream f32 weights once,
// convert to bf16 into LDS, drive v_wmma_f32_16x16x32_bf16 (f32 accum).
// A (activation) tiles are bf16 in global -> copied via GLOBAL_LOAD_ASYNC_TO_LDS_B128
// (ASYNCcnt) into double-buffered LDS, overlapped with the previous tile's WMMAs.

#define BATCH   512
#define DIM     4096
#define NLAYERS 3
#define NCLUST  10
#define SPARSITY 0.2f

#define BM  128
#define BN  128
#define BK  32
#define LDA 40   // LDS row stride (elements): 80B rows -> 16B aligned, bank-rotating

typedef __attribute__((ext_vector_type(16))) __bf16 v16bf;
typedef __attribute__((ext_vector_type(8)))  float  v8f;

union FragAB {
  v16bf   v;
  uint4   q[2];
  uint32_t u[8];
};

__device__ __forceinline__ uint16_t f32_to_bf16(float f) {
  union { __bf16 h; uint16_t u; } c;
  c.h = (__bf16)f;                      // native cvt (RNE), ideally packs to v_cvt_pk
  return c.u;
}

__device__ __forceinline__ float hash_uniform(uint32_t a) {
  a ^= a >> 16; a *= 0x7feb352dU;
  a ^= a >> 15; a *= 0x846ca68bU;
  a ^= a >> 16;
  return (float)(a >> 8) * (1.0f / 16777216.0f);
}

// async global->LDS 16-byte copy, one instruction, tracked by ASYNCcnt
__device__ __forceinline__ void async_b128(uint32_t lds_addr, uint64_t gaddr) {
  asm volatile("global_load_async_to_lds_b128 %0, %1, off"
               :: "v"(lds_addr), "v"(gaddr) : "memory");
}
__device__ __forceinline__ void wait_async0() {
  asm volatile("s_wait_asynccnt 0x0" ::: "memory");
}

// x (f32) -> Xr (bf16), Xi = 0
__global__ void bio_prep_kernel(const float* __restrict__ x,
                                uint16_t* __restrict__ xr,
                                uint16_t* __restrict__ xi) {
  int i = blockIdx.x * blockDim.x + threadIdx.x;
  if (i < BATCH * DIM) {
    xr[i] = f32_to_bf16(x[i]);
    xi[i] = 0;
  }
}

// One layer: (Xr + i*Xi) @ (Wr + i*Wi), fused activation + sparsity mask.
// 256 threads = 8 waves; 128x128 block tile; each wave: 32x64 (2x4 WMMA tiles x {Re,Im}).
__global__ __launch_bounds__(256)
void bio_cgemm_layer(const uint16_t* __restrict__ Xr, const uint16_t* __restrict__ Xi,
                     const float* __restrict__ Wr,  const float* __restrict__ Wi,
                     const float* __restrict__ cw,  const int* __restrict__ asg,
                     const float* __restrict__ beta, const float* __restrict__ alpha,
                     int layer, int is_last,
                     uint16_t* __restrict__ Yr, uint16_t* __restrict__ Yi,
                     float* __restrict__ out) {
  __shared__ uint16_t sAr[2][BM * LDA];   // async-filled, double-buffered
  __shared__ uint16_t sAi[2][BM * LDA];
  __shared__ uint16_t sBr[BN * LDA];      // stored transposed: [n][k]
  __shared__ uint16_t sBi[BN * LDA];

  const int tid   = threadIdx.x;
  const int lane  = tid & 31;
  const int wid   = tid >> 5;
  const int waveM = wid >> 1;             // 0..3 -> 32-row slice
  const int waveN = wid & 1;              // 0..1 -> 64-col slice
  const int m0    = blockIdx.y * BM;
  const int n0    = blockIdx.x * BN;

  // per-thread A-copy geometry: 16 bf16 (2 x b128) per matrix per tile
  const int ar_row = tid >> 1;
  const int ar_col = (tid & 1) * 16;
  const size_t a_gbase = (size_t)(m0 + ar_row) * DIM + ar_col;
  const uint32_t a_lbase = (uint32_t)(ar_row * LDA + ar_col);

  v8f accR[2][4], accI[2][4];
#pragma unroll
  for (int mt = 0; mt < 2; ++mt)
#pragma unroll
    for (int nt = 0; nt < 4; ++nt) { accR[mt][nt] = (v8f)(0.0f); accI[mt][nt] = (v8f)(0.0f); }

  // ---- prologue: kick off async A copies for tile 0 into buffer 0
  {
    uint32_t dr = (uint32_t)(uintptr_t)&sAr[0][a_lbase];
    uint32_t di = (uint32_t)(uintptr_t)&sAi[0][a_lbase];
    async_b128(dr,      (uint64_t)(Xr + a_gbase));
    async_b128(dr + 16, (uint64_t)(Xr + a_gbase + 8));
    async_b128(di,      (uint64_t)(Xi + a_gbase));
    async_b128(di + 16, (uint64_t)(Xi + a_gbase + 8));
  }

  const int NT = DIM / BK;
  for (int t = 0; t < NT; ++t) {
    const int p  = t & 1;
    const int kt = t * BK;

    __syncthreads();  // all waves finished compute(t-1): sB free to restage

    // ---- Stage B: f32 weights rows kt..kt+31, cols n0..n0+127; cvt->bf16,
    //      transpose into sB[n][k] so fragment reads are contiguous b128s.
    {
      int n  = tid & 127;
      int kq = (tid >> 7) * 4;            // 0 or 4
#pragma unroll
      for (int pp = 0; pp < 4; ++pp) {
        int kb = pp * 8 + kq;             // 0..28 step 4
        const float* wr = Wr + (size_t)(kt + kb) * DIM + n0 + n;
        const float* wi = Wi + (size_t)(kt + kb) * DIM + n0 + n;
        uint32_t pr0 = (uint32_t)f32_to_bf16(wr[0])       | ((uint32_t)f32_to_bf16(wr[DIM])     << 16);
        uint32_t pr1 = (uint32_t)f32_to_bf16(wr[2 * DIM]) | ((uint32_t)f32_to_bf16(wr[3 * DIM]) << 16);
        uint32_t pi0 = (uint32_t)f32_to_bf16(wi[0])       | ((uint32_t)f32_to_bf16(wi[DIM])     << 16);
        uint32_t pi1 = (uint32_t)f32_to_bf16(wi[2 * DIM]) | ((uint32_t)f32_to_bf16(wi[3 * DIM]) << 16);
        *(uint2*)&sBr[n * LDA + kb] = make_uint2(pr0, pr1);
        *(uint2*)&sBi[n * LDA + kb] = make_uint2(pi0, pi1);
      }
    }

    // Prefetch next K-slice of weights into cache (global_prefetch_b8)
    if (kt + BK < DIM) {
      int n    = tid & 127;
      int krow = (tid >> 7) * 16;
      __builtin_prefetch(Wr + (size_t)(kt + BK + krow) * DIM + n0 + n, 0, 1);
      __builtin_prefetch(Wi + (size_t)(kt + BK + krow) * DIM + n0 + n, 0, 1);
    }

    wait_async0();      // A(tile t) fully landed in sA*[p]
    __syncthreads();    // B staged + async A visible to all waves

    // ---- issue async A copies for tile t+1 into the other buffer
    //      (safe: sA*[1-p] was last read in compute(t-1), all waves are past it)
    if (t + 1 < NT) {
      uint32_t dr = (uint32_t)(uintptr_t)&sAr[1 - p][a_lbase];
      uint32_t di = (uint32_t)(uintptr_t)&sAi[1 - p][a_lbase];
      size_t g = a_gbase + kt + BK;
      async_b128(dr,      (uint64_t)(Xr + g));
      async_b128(dr + 16, (uint64_t)(Xr + g + 8));
      async_b128(di,      (uint64_t)(Xi + g));
      async_b128(di + 16, (uint64_t)(Xi + g + 8));
    }

    // ---- A fragments (ISA 16-bit A layout: lanes 0-15 K{0-7,16-23}, lanes 16-31 K{8-15,24-31})
    FragAB ar[2], ai[2], ain[2];
    {
      int khalf = (lane >> 4) * 8;
#pragma unroll
      for (int mt = 0; mt < 2; ++mt) {
        int m = waveM * 32 + mt * 16 + (lane & 15);
        ar[mt].q[0] = *(const uint4*)&sAr[p][m * LDA + khalf];
        ar[mt].q[1] = *(const uint4*)&sAr[p][m * LDA + khalf + 16];
        ai[mt].q[0] = *(const uint4*)&sAi[p][m * LDA + khalf];
        ai[mt].q[1] = *(const uint4*)&sAi[p][m * LDA + khalf + 16];
#pragma unroll
        for (int j = 0; j < 8; ++j) ain[mt].u[j] = ai[mt].u[j] ^ 0x80008000u;  // -Ai
      }
    }

    // ---- B fragments (lanes 0-15: K{0-7,16-23} of column N; lanes 16-31: K{8-15,24-31}) + WMMA
    {
      int khalf2 = (lane >> 4) * 8;
#pragma unroll
      for (int nt = 0; nt < 4; ++nt) {
        FragAB br, bi;
        int n = waveN * 64 + nt * 16 + (lane & 15);
        br.q[0] = *(const uint4*)&sBr[n * LDA + khalf2];
        br.q[1] = *(const uint4*)&sBr[n * LDA + khalf2 + 16];
        bi.q[0] = *(const uint4*)&sBi[n * LDA + khalf2];
        bi.q[1] = *(const uint4*)&sBi[n * LDA + khalf2 + 16];
#pragma unroll
        for (int mt = 0; mt < 2; ++mt) {
          // Re += Ar*Br - Ai*Bi ; Im += Ar*Bi + Ai*Br   (bf16 WMMA has no A/B NEG -> sign-xor'd Ai)
          accR[mt][nt] = __builtin_amdgcn_wmma_f32_16x16x32_bf16(false, ar[mt].v,  false, br.v, (short)0, accR[mt][nt], false, false);
          accR[mt][nt] = __builtin_amdgcn_wmma_f32_16x16x32_bf16(false, ain[mt].v, false, bi.v, (short)0, accR[mt][nt], false, false);
          accI[mt][nt] = __builtin_amdgcn_wmma_f32_16x16x32_bf16(false, ar[mt].v,  false, bi.v, (short)0, accI[mt][nt], false, false);
          accI[mt][nt] = __builtin_amdgcn_wmma_f32_16x16x32_bf16(false, ai[mt].v,  false, br.v, (short)0, accI[mt][nt], false, false);
        }
      }
    }
  }

  // ---- Epilogue: sigmoid(beta*Re)^alpha at original phase, cluster sparsity mask
  float cwv[NCLUST];
  float cmax = -1e30f;
#pragma unroll
  for (int c = 0; c < NCLUST; ++c) { cwv[c] = cw[c]; cmax = fmaxf(cmax, cwv[c]); }
  float csum = 0.0f;
#pragma unroll
  for (int c = 0; c < NCLUST; ++c) { cwv[c] = __expf(cwv[c] - cmax); csum += cwv[c]; }
  const float pscale = SPARSITY / csum;
  const float bet = beta[layer];
  const float alp = alpha[layer];

#pragma unroll
  for (int nt = 0; nt < 4; ++nt) {
    int N = n0 + waveN * 64 + nt * 16 + (lane & 15);
    float prob = cwv[asg[N]] * pscale;
#pragma unroll
    for (int mt = 0; mt < 2; ++mt) {
#pragma unroll
      for (int r = 0; r < 8; ++r) {
        int M = m0 + waveM * 32 + mt * 16 + (lane >> 4) * 8 + r;
        float orr = accR[mt][nt][r];
        float oii = accI[mt][nt][r];
        float act = __powf(1.0f / (1.0f + __expf(-bet * orr)), alp);
        float m2  = orr * orr + oii * oii;
        float inv = (m2 > 0.0f) ? rsqrtf(m2) : 0.0f;
        float re  = act * orr * inv;
        float im  = act * oii * inv;
        // stateless hash RNG standing in for jax.random Bernoulli(p)
        uint32_t h = (uint32_t)(M * DIM + N) * 0x9E3779B9u + (uint32_t)(layer + 1) * 0x85EBCA6Bu;
        float msk = (hash_uniform(h) < prob) ? 1.0f : 0.0f;
        re *= msk; im *= msk;
        size_t idx = (size_t)M * DIM + N;
        if (is_last) {
          out[2 * idx]     = re;    // complex64 interleaved
          out[2 * idx + 1] = im;
        } else {
          Yr[idx] = f32_to_bf16(re);
          Yi[idx] = f32_to_bf16(im);
        }
      }
    }
  }
}

extern "C" void kernel_launch(void* const* d_in, const int* in_sizes, int n_in,
                              void* d_out, int out_size, void* d_ws, size_t ws_size,
                              hipStream_t stream) {
  (void)in_sizes; (void)n_in; (void)out_size; (void)ws_size;
  // setup_inputs() dict order:
  // 0:x 1:W0r 2:W0i 3:cw0 4:asg0 5:W1r 6:W1i 7:cw1 8:asg1 9:W2r 10:W2i 11:cw2 12:asg2 13:beta 14:alpha
  const float* x      = (const float*)d_in[0];
  const float* Wr[3]  = { (const float*)d_in[1], (const float*)d_in[5], (const float*)d_in[9]  };
  const float* Wi[3]  = { (const float*)d_in[2], (const float*)d_in[6], (const float*)d_in[10] };
  const float* cwp[3] = { (const float*)d_in[3], (const float*)d_in[7], (const float*)d_in[11] };
  const int*   asg[3] = { (const int*)d_in[4],   (const int*)d_in[8],   (const int*)d_in[12]   };
  const float* beta   = (const float*)d_in[13];
  const float* alpha  = (const float*)d_in[14];

  const size_t act_elems = (size_t)BATCH * DIM;
  uint16_t* bufs = (uint16_t*)d_ws;
  uint16_t* Xr = bufs;
  uint16_t* Xi = bufs + act_elems;
  uint16_t* Yr = bufs + 2 * act_elems;
  uint16_t* Yi = bufs + 3 * act_elems;

  bio_prep_kernel<<<(BATCH * DIM + 255) / 256, 256, 0, stream>>>(x, Xr, Xi);

  dim3 grid(DIM / BN, BATCH / BM);
  for (int L = 0; L < NLAYERS; ++L) {
    int last = (L == NLAYERS - 1);
    bio_cgemm_layer<<<grid, 256, 0, stream>>>(Xr, Xi, Wr[L], Wi[L], cwp[L], asg[L],
                                              beta, alpha, L, last, Yr, Yi, (float*)d_out);
    uint16_t* t;
    t = Xr; Xr = Yr; Yr = t;
    t = Xi; Xi = Yi; Yi = t;
  }
}
